// FragMoE_Full_73572789781173
// MI455X (gfx1250) — matrix operand order repaired and checked by reading, not compiled, not run on hardware
//
#include <hip/hip_runtime.h>
#include <hip/hip_bf16.h>

typedef __attribute__((ext_vector_type(16))) _Float16 v16h;
typedef __attribute__((ext_vector_type(8)))  _Float16 v8h;
typedef __attribute__((ext_vector_type(4)))  _Float16 v4h;
typedef __attribute__((ext_vector_type(8)))  float    v8f;

#define NNODES 100000
#define NEDGES 600000
#define NFRAGS 50000
#define NMOLS  10000
#define PADN   100096   // 782 tiles * 128 rows
#define PADF   50048    // 391 tiles * 128 rows

// ---------------------------------------------------------------------------
// WMMA helpers (CDNA5 16x16x32 f16, f32 accumulate)
// ---------------------------------------------------------------------------
__device__ __forceinline__ v8f wmma_f16(v16h a, v16h b, v8f c) {
    return __builtin_amdgcn_wmma_f32_16x16x32_f16(
        /*neg_a=*/false, a, /*neg_b=*/false, b,
        /*c_mod=*/(short)0, c, /*reuse_a=*/false, /*reuse_b=*/false);
}

// A fragment: prow points at this lane's row (row = m0 + (lane&15)), row-major.
// 16-bit A layout: lanes 0-15 (g=0): K = e (e<8), K = e+8 (e>=8)
//                  lanes 16-31 (g=1): K = e+8 (e<8), K = e+16 (e>=8)
__device__ __forceinline__ v16h make_a_frag(const _Float16* prow, int k0, int g) {
    v8h lo = *(const v8h*)(prow + k0 + 8 * g);
    v8h hi = *(const v8h*)(prow + k0 + 16 + 8 * g);
    v16h a;
#pragma unroll
    for (int i = 0; i < 8; ++i) { a[i] = lo[i]; a[8 + i] = hi[i]; }
    return a;
}

// B fragment from f16 weights stored TRANSPOSED: Wt[N][K], K-stride = 128.
// Lane L: N = L&15, g = L>>4, element e -> K = k0 + 8*g + e (e<8), k0 + 16 + 8*g + (e-8) (e>=8).
__device__ __forceinline__ v16h make_b_frag(const _Float16* wt, int n0, int k0, int lane) {
    int n = lane & 15, g = lane >> 4;
    const _Float16* p = wt + (size_t)(n0 + n) * 128 + k0 + 8 * g;
    v8h lo = *(const v8h*)(p);
    v8h hi = *(const v8h*)(p + 16);
    v16h b;
#pragma unroll
    for (int i = 0; i < 8; ++i) { b[i] = lo[i]; b[8 + i] = hi[i]; }
    return b;
}

// ---------------------------------------------------------------------------
// Small utility kernels
// ---------------------------------------------------------------------------
__global__ void copy4_k(const float4* __restrict__ s, float4* __restrict__ d, int n4) {
    int i = blockIdx.x * 256 + threadIdx.x;
    if (i < n4) d[i] = s[i];
}

template <int DIM>
__global__ void scatter_add_k(const float* __restrict__ feat, const int* __restrict__ esrc,
                              const int* __restrict__ edst, float* __restrict__ out, int nedges) {
    int i = blockIdx.x * 256 + threadIdx.x;
    if (i >= nedges * DIM) return;
    int e = i / DIM, d = i - e * DIM;   // DIM is a compile-time constant -> shifts for 128
    atomicAdd(&out[(size_t)edst[e] * DIM + d], feat[(size_t)esrc[e] * DIM + d]);
}

__global__ void bn_prep_k(const float* g, const float* b, const float* m, const float* v,
                          float* scale, float* shift, int n) {
    int i = threadIdx.x;
    if (i < n) {
        float sc = g[i] * rsqrtf(v[i] + 1e-5f);
        scale[i] = sc;
        shift[i] = b[i] - m[i] * sc;
    }
}

// W[b][k][n] (f32) -> Wt[b][n][k] (f16)
__global__ void conv_t_k(const float* __restrict__ W, _Float16* __restrict__ Wt,
                         int K, int N, int total) {
    int i = blockIdx.x * 256 + threadIdx.x;
    if (i >= total) return;
    int b = i / (K * N);
    int rem = i - b * K * N;
    int k = rem / N, n = rem - k * N;
    Wt[(size_t)b * K * N + (size_t)n * K + k] = (_Float16)W[i];
}

__global__ void count_seg_k(const int* __restrict__ seg, float* __restrict__ cnt, int n) {
    int i = blockIdx.x * 256 + threadIdx.x;
    if (i < n) atomicAdd(&cnt[seg[i]], 1.0f);
}

__global__ void pool_sum_k(const float* __restrict__ H, const int* __restrict__ batch,
                           float* __restrict__ fsum, int nnodes) {
    int i = blockIdx.x * 256 + threadIdx.x;
    if (i >= nnodes * 128) return;
    int n = i >> 7, d = i & 127;
    atomicAdd(&fsum[(size_t)batch[n] * 128 + d], H[i]);
}

__global__ void divide_k(float* __restrict__ fsum, const float* __restrict__ cnt,
                         _Float16* __restrict__ frag16, int nfr) {
    int i = blockIdx.x * 256 + threadIdx.x;
    if (i >= nfr * 128) return;
    int f = i >> 7;
    float c = cnt[f]; c = c > 1.0f ? c : 1.0f;
    float v = fsum[i] / c;
    fsum[i] = v;              // frag_emb in place (f32)
    frag16[i] = (_Float16)v;  // f16 copy for WMMA experts
}

__global__ void finalize_k(const float* msum, const float* mcnt, float* out, int n) {
    int i = blockIdx.x * 256 + threadIdx.x;
    if (i >= n) return;
    float c = mcnt[i];
    out[i] = (c > 0.0f) ? msum[i] / (c > 1.0f ? c : 1.0f) : 0.0f;
}

// ---------------------------------------------------------------------------
// Layer 0 FC1 (K=9, too small for WMMA): scalar, writes f16 activations
// ---------------------------------------------------------------------------
__global__ void l0_fc1_k(const float* __restrict__ Z9, const float* __restrict__ W1,
                         const float* __restrict__ b1, _Float16* __restrict__ Y, int nnodes) {
    int i = blockIdx.x * 256 + threadIdx.x;
    if (i >= nnodes * 128) return;
    int n = i >> 7, o = i & 127;
    float acc = b1[o];
#pragma unroll
    for (int k = 0; k < 9; ++k) acc += Z9[n * 9 + k] * W1[k * 128 + o];
    Y[i] = (_Float16)(acc > 0.0f ? acc : 0.0f);
}

// ---------------------------------------------------------------------------
// Single WMMA GEMM (K=128, N=128) + bias + BN + ReLU; A read directly from
// global f16 (row-major [*,128], padded to a 128-row multiple -> no guards).
// ---------------------------------------------------------------------------
__global__ __launch_bounds__(256) void gemm_bn_k(
    const _Float16* __restrict__ A, const _Float16* __restrict__ Wt,
    const float* __restrict__ bias, const float* __restrict__ bn_s,
    const float* __restrict__ bn_sh, float* __restrict__ H) {
    int row0 = blockIdx.x * 128;
    int wave = threadIdx.x >> 5, lane = threadIdx.x & 31;
    int m = lane & 15, g = lane >> 4, nn = lane & 15;
    int m0 = wave * 16;
    const _Float16* prow = A + (size_t)(row0 + m0 + m) * 128;
    v16h afr[4];
#pragma unroll
    for (int kc = 0; kc < 4; ++kc) afr[kc] = make_a_frag(prow, kc * 32, g);
#pragma unroll
    for (int t = 0; t < 8; ++t) {
        v8f acc = {};
#pragma unroll
        for (int kc = 0; kc < 4; ++kc)
            acc = wmma_f16(afr[kc], make_b_frag(Wt, t * 16, kc * 32, lane), acc);
        int col = t * 16 + nn;
        float bs = bias[col], s = bn_s[col], sh = bn_sh[col];
#pragma unroll
        for (int r = 0; r < 8; ++r) {
            float v = (acc[r] + bs) * s + sh;
            H[(size_t)(row0 + m0 + r + 8 * g) * 128 + col] = v > 0.0f ? v : 0.0f;
        }
    }
}

// ---------------------------------------------------------------------------
// Fused GIN MLP (layers 1,2): Z(f32) -> relu(Z@W1+b1) -> BN(·@W2+b2) -> relu
// Single 34KB LDS tile; each wave overwrites only its own 16 rows between the
// two GEMMs (GEMM1 results held entirely in accumulator registers).
// Z and H padded to 128-row multiples -> no bounds checks anywhere.
// ---------------------------------------------------------------------------
__global__ __launch_bounds__(256) void gin_mlp_k(
    const float* __restrict__ Z, const _Float16* __restrict__ W1t,
    const float* __restrict__ b1, const _Float16* __restrict__ W2t,
    const float* __restrict__ b2, const float* __restrict__ bn_s,
    const float* __restrict__ bn_sh, float* __restrict__ H) {
    __shared__ _Float16 sA[128 * 136];  // 34816 B, rows 16B-aligned (136*2 = 272)
    int row0 = blockIdx.x * 128;
    int tid = threadIdx.x;
    // float4 loads -> packed 4xf16 LDS stores (ds_store_b64)
    for (int idx = tid; idx < 128 * 32; idx += 256) {
        int r = idx >> 5, c4 = (idx & 31) * 4;
        float4 v = *(const float4*)(Z + (size_t)(row0 + r) * 128 + c4);
        v4h hv = {(_Float16)v.x, (_Float16)v.y, (_Float16)v.z, (_Float16)v.w};
        *(v4h*)(sA + r * 136 + c4) = hv;
    }
    __syncthreads();
    int wave = tid >> 5, lane = tid & 31;
    int m = lane & 15, g = lane >> 4, nn = lane & 15;
    int m0 = wave * 16;
    const _Float16* prow = sA + (m0 + m) * 136;

    v16h afr[4];
#pragma unroll
    for (int kc = 0; kc < 4; ++kc) afr[kc] = make_a_frag(prow, kc * 32, g);

    // GEMM1: keep all 8 N-tiles in registers
    v8f accs[8];
#pragma unroll
    for (int t = 0; t < 8; ++t) {
        v8f acc = {};
#pragma unroll
        for (int kc = 0; kc < 4; ++kc)
            acc = wmma_f16(afr[kc], make_b_frag(W1t, t * 16, kc * 32, lane), acc);
        accs[t] = acc;
    }
    // relu(+bias) -> overwrite own 16 LDS rows in f16
#pragma unroll
    for (int t = 0; t < 8; ++t) {
        int col = t * 16 + nn;
        float bs = b1[col];
#pragma unroll
        for (int r = 0; r < 8; ++r) {
            float v = accs[t][r] + bs;
            sA[(m0 + r + 8 * g) * 136 + col] = (_Float16)(v > 0.0f ? v : 0.0f);
        }
    }
    __syncthreads();
    // GEMM2 + BN + relu -> global
#pragma unroll
    for (int kc = 0; kc < 4; ++kc) afr[kc] = make_a_frag(prow, kc * 32, g);
#pragma unroll
    for (int t = 0; t < 8; ++t) {
        v8f acc = {};
#pragma unroll
        for (int kc = 0; kc < 4; ++kc)
            acc = wmma_f16(afr[kc], make_b_frag(W2t, t * 16, kc * 32, lane), acc);
        int col = t * 16 + nn;
        float bs = b2[col], s = bn_s[col], sh = bn_sh[col];
#pragma unroll
        for (int r = 0; r < 8; ++r) {
            float v = (acc[r] + bs) * s + sh;
            H[(size_t)(row0 + m0 + r + 8 * g) * 128 + col] = v > 0.0f ? v : 0.0f;
        }
    }
}

// ---------------------------------------------------------------------------
// Router: fc1 (128->64, relu) then logits (64->4) + softmax
// ---------------------------------------------------------------------------
__global__ void router_fc1_k(const float* __restrict__ frag, const float* __restrict__ w1,
                             const float* __restrict__ b1, float* __restrict__ h1, int nfr) {
    int i = blockIdx.x * 256 + threadIdx.x;
    if (i >= nfr * 64) return;
    int f = i >> 6, o = i & 63;
    float acc = b1[o];
    const float* fr = frag + (size_t)f * 128;
#pragma unroll 8
    for (int k = 0; k < 128; ++k) acc += fr[k] * w1[k * 64 + o];
    h1[i] = acc > 0.0f ? acc : 0.0f;
}

__global__ void router_soft_k(const float* __restrict__ h1, const float* __restrict__ w2,
                              const float* __restrict__ b2, float* __restrict__ wts, int nfr) {
    int f = blockIdx.x * 256 + threadIdx.x;
    if (f >= nfr) return;
    float l[4] = {b2[0], b2[1], b2[2], b2[3]};
    const float* h = h1 + (size_t)f * 64;
    for (int k = 0; k < 64; ++k) {
        float hv = h[k];
#pragma unroll
        for (int c = 0; c < 4; ++c) l[c] += hv * w2[k * 4 + c];
    }
    float mx = l[0];
#pragma unroll
    for (int c = 1; c < 4; ++c) mx = l[c] > mx ? l[c] : mx;
    float s = 0.0f;
#pragma unroll
    for (int c = 0; c < 4; ++c) { l[c] = __expf(l[c] - mx); s += l[c]; }
    float inv = 1.0f / s;
#pragma unroll
    for (int c = 0; c < 4; ++c) wts[f * 4 + c] = l[c] * inv;
}

// ---------------------------------------------------------------------------
// Experts: per 128-fragment tile, 4 experts, WMMA 128->128 (relu) then
// 128->64 (relu), stage-3 dot with w3 via in-wave shuffles, weighted sum,
// atomic add into molecule sums. F16 padded to 128-row multiple.
// ---------------------------------------------------------------------------
__global__ __launch_bounds__(256) void experts_k(
    const _Float16* __restrict__ F16, const _Float16* __restrict__ ew1t,
    const float* __restrict__ eb1, const _Float16* __restrict__ ew2t,
    const float* __restrict__ eb2, const float* __restrict__ ew3,
    const float* __restrict__ eb3, const float* __restrict__ wts,
    const int* __restrict__ mol, float* __restrict__ msum, int nfr) {
    __shared__ _Float16 sY[128 * 136];  // e1 activations, 34816 B
    int row0 = blockIdx.x * 128;
    int tid = threadIdx.x, wave = tid >> 5, lane = tid & 31;
    int m = lane & 15, g = lane >> 4, nn = lane & 15;
    int m0 = wave * 16;
    const _Float16* prow = F16 + (size_t)(row0 + m0 + m) * 128;
    v16h afr[4];
#pragma unroll
    for (int kc = 0; kc < 4; ++kc) afr[kc] = make_a_frag(prow, kc * 32, g);

    float wacc[8];
#pragma unroll
    for (int r = 0; r < 8; ++r) wacc[r] = 0.0f;

    for (int e = 0; e < 4; ++e) {
        const _Float16* w1 = ew1t + (size_t)e * 128 * 128;
        const _Float16* w2 = ew2t + (size_t)e * 64 * 128;
        // GEMM1: frag @ e_w1 -> relu -> own 16 rows of sY (f16)
#pragma unroll
        for (int t = 0; t < 8; ++t) {
            v8f acc = {};
#pragma unroll
            for (int kc = 0; kc < 4; ++kc)
                acc = wmma_f16(afr[kc], make_b_frag(w1, t * 16, kc * 32, lane), acc);
            int col = t * 16 + nn;
            float bs = eb1[e * 128 + col];
#pragma unroll
            for (int r = 0; r < 8; ++r) {
                float v = acc[r] + bs;
                sY[(m0 + r + 8 * g) * 136 + col] = (_Float16)(v > 0.0f ? v : 0.0f);
            }
        }
        __syncthreads();
        const _Float16* pyrow = sY + (m0 + m) * 136;
        v16h yfr[4];
#pragma unroll
        for (int kc = 0; kc < 4; ++kc) yfr[kc] = make_a_frag(pyrow, kc * 32, g);
        // GEMM2 (N=64) + relu, fused with w3 dot (per-lane partial over n)
        float part[8];
#pragma unroll
        for (int r = 0; r < 8; ++r) part[r] = 0.0f;
#pragma unroll
        for (int t = 0; t < 4; ++t) {
            v8f acc = {};
#pragma unroll
            for (int kc = 0; kc < 4; ++kc)
                acc = wmma_f16(yfr[kc], make_b_frag(w2, t * 16, kc * 32, lane), acc);
            int col = t * 16 + nn;
            float bs = eb2[e * 64 + col];
            float w3 = ew3[e * 64 + col];
#pragma unroll
            for (int r = 0; r < 8; ++r) {
                float v = acc[r] + bs;
                v = v > 0.0f ? v : 0.0f;
                part[r] += v * w3;
            }
        }
        // reduce across the 16 lanes of this half-wave (xor 1,2,4,8 stay in-group)
#pragma unroll
        for (int r = 0; r < 8; ++r) {
            float s = part[r];
            s += __shfl_xor(s, 1, 32);
            s += __shfl_xor(s, 2, 32);
            s += __shfl_xor(s, 4, 32);
            s += __shfl_xor(s, 8, 32);
            int mr = row0 + m0 + r + 8 * g;
            if (nn == 0 && mr < nfr) {
                float val = s + eb3[e];
                wacc[r] += wts[mr * 4 + e] * val;
            }
        }
        __syncthreads();
    }
#pragma unroll
    for (int r = 0; r < 8; ++r) {
        int mr = row0 + m0 + r + 8 * g;
        if (nn == 0 && mr < nfr) atomicAdd(&msum[mol[mr]], wacc[r]);
    }
}

// ---------------------------------------------------------------------------
// Host-side launch
// ---------------------------------------------------------------------------
extern "C" void kernel_launch(void* const* d_in, const int* in_sizes, int n_in,
                              void* d_out, int out_size, void* d_ws, size_t ws_size,
                              hipStream_t stream) {
    const float* x = (const float*)d_in[0];
    const int* edges = (const int*)d_in[1];
    const int* esrc = edges;
    const int* edst = edges + NEDGES;
    const int* batch = (const int*)d_in[2];
    const int* mol = (const int*)d_in[3];
    const float *gw1[3], *gb1[3], *gw2[3], *gb2[3], *bng[3], *bnb[3], *bnm[3], *bnv[3];
    for (int i = 0; i < 3; ++i) {
        int b = 4 + i * 8;
        gw1[i] = (const float*)d_in[b + 0]; gb1[i] = (const float*)d_in[b + 1];
        gw2[i] = (const float*)d_in[b + 2]; gb2[i] = (const float*)d_in[b + 3];
        bng[i] = (const float*)d_in[b + 4]; bnb[i] = (const float*)d_in[b + 5];
        bnm[i] = (const float*)d_in[b + 6]; bnv[i] = (const float*)d_in[b + 7];
    }
    const float* rw1 = (const float*)d_in[28]; const float* rb1 = (const float*)d_in[29];
    const float* rw2 = (const float*)d_in[30]; const float* rb2 = (const float*)d_in[31];
    const float* ew1 = (const float*)d_in[32]; const float* eb1 = (const float*)d_in[33];
    const float* ew2 = (const float*)d_in[34]; const float* eb2 = (const float*)d_in[35];
    const float* ew3 = (const float*)d_in[36]; const float* eb3 = (const float*)d_in[37];

    // -------- workspace layout (row-padded activation buffers) --------
    char* ws = (char*)d_ws;
    float* bufZ = (float*)ws;                         // PADN*128*4 = 51,249,152 B
    float* bufH = (float*)(ws + 51249152);            // PADN*128*4
    char* bufC = ws + 102498304;                      // max(Y0 f16 padded, fsum f32)
    _Float16* Y0 = (_Float16*)bufC;                   // PADN*128*2 = 25,624,576 B
    float* fsum = (float*)bufC;                       // NFRAGS*128*4 = 25,600,000 B
    char* sm = ws + 128122880;
    _Float16* w2t0 = (_Float16*)sm; sm += 32768;
    _Float16* w1t1 = (_Float16*)sm; sm += 32768;
    _Float16* w2t1 = (_Float16*)sm; sm += 32768;
    _Float16* w1t2 = (_Float16*)sm; sm += 32768;
    _Float16* w2t2 = (_Float16*)sm; sm += 32768;
    _Float16* ew1t = (_Float16*)sm; sm += 131072;
    _Float16* ew2t = (_Float16*)sm; sm += 65536;
    float* bns = (float*)sm; sm += 3 * 256 * sizeof(float);  // [layer]{scale128, shift128}
    float* fcnt = (float*)sm; sm += 200192;
    float* msum = (float*)sm; sm += 40192;
    float* mcnt = (float*)sm; sm += 40192;
    // aliases inside bufZ (valid after layer-2 GEMMs consume Z)
    _Float16* frag16 = (_Float16*)bufZ;               // PADF*128*2 = 12,812,288 B
    float* h1r = (float*)(ws + 16000000);             // 12.8 MB
    float* wts = (float*)(ws + 41600000);             // 0.8 MB

    auto NB = [](long long n) { return (unsigned)((n + 255) / 256); };
    const _Float16* w1t[3] = {nullptr, w1t1, w1t2};
    const _Float16* w2t[3] = {w2t0, w2t1, w2t2};

    // -------- weight prep --------
    for (int l = 0; l < 3; ++l)
        bn_prep_k<<<1, 128, 0, stream>>>(bng[l], bnb[l], bnm[l], bnv[l],
                                         bns + l * 256, bns + l * 256 + 128, 128);
    conv_t_k<<<NB(128 * 128), 256, 0, stream>>>(gw2[0], w2t0, 128, 128, 128 * 128);
    conv_t_k<<<NB(128 * 128), 256, 0, stream>>>(gw1[1], w1t1, 128, 128, 128 * 128);
    conv_t_k<<<NB(128 * 128), 256, 0, stream>>>(gw2[1], w2t1, 128, 128, 128 * 128);
    conv_t_k<<<NB(128 * 128), 256, 0, stream>>>(gw1[2], w1t2, 128, 128, 128 * 128);
    conv_t_k<<<NB(128 * 128), 256, 0, stream>>>(gw2[2], w2t2, 128, 128, 128 * 128);
    conv_t_k<<<NB(4 * 128 * 128), 256, 0, stream>>>(ew1, ew1t, 128, 128, 4 * 128 * 128);
    conv_t_k<<<NB(4 * 128 * 64), 256, 0, stream>>>(ew2, ew2t, 128, 64, 4 * 128 * 64);

    // -------- layer 0 (D_IN=9) --------
    copy4_k<<<NB(NNODES * 9 / 4), 256, 0, stream>>>((const float4*)x, (float4*)bufZ,
                                                    NNODES * 9 / 4);
    scatter_add_k<9><<<NB((long long)NEDGES * 9), 256, 0, stream>>>(x, esrc, edst, bufZ, NEDGES);
    l0_fc1_k<<<NB(NNODES * 128), 256, 0, stream>>>(bufZ, gw1[0], gb1[0], Y0, NNODES);
    gemm_bn_k<<<PADN / 128, 256, 0, stream>>>(Y0, w2t0, gb2[0], bns, bns + 128, bufH);

    // -------- layers 1,2 (D=128, fused WMMA MLP) --------
    for (int l = 1; l < 3; ++l) {
        copy4_k<<<NB(NNODES * 32), 256, 0, stream>>>((const float4*)bufH, (float4*)bufZ,
                                                     NNODES * 32);
        scatter_add_k<128><<<NB((long long)NEDGES * 128), 256, 0, stream>>>(bufH, esrc, edst,
                                                                            bufZ, NEDGES);
        gin_mlp_k<<<PADN / 128, 256, 0, stream>>>(bufZ, w1t[l], gb1[l], w2t[l], gb2[l],
                                                  bns + l * 256, bns + l * 256 + 128, bufH);
    }

    // -------- fragment mean pool --------
    hipMemsetAsync(fsum, 0, (size_t)NFRAGS * 128 * 4, stream);
    hipMemsetAsync(fcnt, 0, (size_t)NFRAGS * 4, stream);
    pool_sum_k<<<NB(NNODES * 128), 256, 0, stream>>>(bufH, batch, fsum, NNODES);
    count_seg_k<<<NB(NNODES), 256, 0, stream>>>(batch, fcnt, NNODES);
    divide_k<<<NB(NFRAGS * 128), 256, 0, stream>>>(fsum, fcnt, frag16, NFRAGS);

    // -------- router --------
    router_fc1_k<<<NB(NFRAGS * 64), 256, 0, stream>>>(fsum, rw1, rb1, h1r, NFRAGS);
    router_soft_k<<<NB(NFRAGS), 256, 0, stream>>>(h1r, rw2, rb2, wts, NFRAGS);

    // -------- experts + molecule mean --------
    hipMemsetAsync(msum, 0, (size_t)NMOLS * 4, stream);
    hipMemsetAsync(mcnt, 0, (size_t)NMOLS * 4, stream);
    count_seg_k<<<NB(NFRAGS), 256, 0, stream>>>(mol, mcnt, NFRAGS);
    experts_k<<<PADF / 128, 256, 0, stream>>>(frag16, ew1t, eb1, ew2t, eb2, ew3, eb3,
                                              wts, mol, msum, NFRAGS);
    finalize_k<<<NB(NMOLS), 256, 0, stream>>>(msum, mcnt, (float*)d_out, NMOLS);
}
